// GAT_37237366456904
// MI455X (gfx1250) — matrix operand order repaired and checked by reading, not compiled, not run on hardware
//
#include <hip/hip_runtime.h>
#include <hip/hip_bf16.h>
#include <math.h>

// ---------------- constants from the reference ----------------
#define HCW   256   // H*C concat width
#define NHEAD 8
#define CHAN  32
#define NOUT  10
#define SLOPE 0.2f

typedef _Float16 v16h __attribute__((ext_vector_type(16)));
typedef _Float16 v8h  __attribute__((ext_vector_type(8)));
typedef float    v8f  __attribute__((ext_vector_type(8)));

// ---------------- helpers ----------------
__device__ __forceinline__ float atomicMaxFloat(float* addr, float val) {
  // sign-aware trick: positive floats order like ints, negative like reversed uints
  if (val >= 0.0f)
    return __int_as_float(atomicMax((int*)addr, __float_as_int(val)));
  else
    return __uint_as_float(atomicMin((unsigned int*)addr, __float_as_uint(val)));
}

__device__ __forceinline__ float lrelu(float v) { return v > 0.0f ? v : SLOPE * v; }

// ---------------- small utility kernels ----------------
__global__ void k_fill(float* __restrict__ p, float v, long n) {
  long i = (long)blockIdx.x * blockDim.x + threadIdx.x;
  if (i < n) p[i] = v;
}

__global__ void k_convert_half(const float* __restrict__ src, _Float16* __restrict__ dst, long n) {
  long i = (long)blockIdx.x * blockDim.x + threadIdx.x;
  if (i < n) dst[i] = (_Float16)src[i];
}

// W is [K, HCW] row-major -> Wt is [HCW, K] row-major (column-major B, contiguous along K)
__global__ void k_transpose_half(const float* __restrict__ W, _Float16* __restrict__ Wt, int K) {
  int i = blockIdx.x * blockDim.x + threadIdx.x;
  if (i >= K * HCW) return;
  int r = i / HCW, c = i % HCW;
  Wt[(size_t)c * K + r] = (_Float16)W[i];
}

// ---------------- WMMA GEMM: C[M,HCW] = A[M,K] (f16) x B[K,HCW] (f16, given as Bt[HCW,K]) ----------------
// block = 128 threads = 4 waves; each wave computes a 16x64 strip of C.
__global__ __launch_bounds__(128)
void k_gemm_wmma(const _Float16* __restrict__ A, const _Float16* __restrict__ Bt,
                 float* __restrict__ C, int M, int K) {
  const int lane = threadIdx.x & 31;
  const int wave = threadIdx.x >> 5;
  const int row0 = blockIdx.x * 16;
  const int col0 = wave * 64;
  const int r    = lane & 15;   // row within tile (A) / col within tile (B)
  const int hi   = lane >> 4;   // lane-half selector

  v8f acc[4];
#pragma unroll
  for (int t = 0; t < 4; ++t) acc[t] = (v8f){0,0,0,0,0,0,0,0};

  int rowA = row0 + r; if (rowA >= M) rowA = M - 1;   // safe (M=50000 is 16-aligned anyway)
  const _Float16* __restrict__ arow = A + (size_t)rowA * K;

  for (int kk = 0; kk < K; kk += 32) {
    // A fragment, 16x32 f16, contiguous k order (same order as B, not the ISA table order):
    // lane = hi*16+r holds row r, K = kk + hi*16 + [0..15] contiguous
    const int kbase = kk + hi * 16;
    v8h alo = *(const v8h*)(arow + kbase);
    v8h ahi = *(const v8h*)(arow + kbase + 8);
    v16h af;
#pragma unroll
    for (int i = 0; i < 8; ++i) { af[i] = alo[i]; af[8 + i] = ahi[i]; }

    // B fragments: lane holds column (col0+t*16+r), K = kk + hi*16 + [0..15] contiguous
#pragma unroll
    for (int t = 0; t < 4; ++t) {
      const _Float16* bp = Bt + (size_t)(col0 + t * 16 + r) * K + kk + hi * 16;
      v16h bf = *(const v16h*)bp;
      acc[t] = __builtin_amdgcn_wmma_f32_16x16x32_f16(
          /*neg_a=*/false, af, /*neg_b=*/false, bf,
          /*c_mod=*/(short)0, acc[t], /*reuse_a=*/false, /*reuse_b=*/false);
    }
  }

  // C/D layout: VGPR v -> row = row0 + v + hi*8, col = col0 + t*16 + r
#pragma unroll
  for (int t = 0; t < 4; ++t)
#pragma unroll
    for (int v = 0; v < 8; ++v) {
      int row = row0 + v + hi * 8;
      if (row < M) C[(size_t)row * HCW + col0 + t * 16 + r] = acc[t][v];
    }
}

// ---------------- GAT attention kernels ----------------
// per-(node,head) source/dest scores: s = <h[n,h,:], a_src[h,:]>, d likewise
__global__ void k_scores(const float* __restrict__ h, const float* __restrict__ asrc,
                         const float* __restrict__ adst, float* __restrict__ s,
                         float* __restrict__ d, int N) {
  int i = blockIdx.x * blockDim.x + threadIdx.x;
  if (i >= N * NHEAD) return;
  int head = i & (NHEAD - 1);
  const float* hp = h + (size_t)(i >> 3) * HCW + head * CHAN;
  const float* ap = asrc + head * CHAN;
  const float* bp = adst + head * CHAN;
  float ss = 0.f, dd = 0.f;
#pragma unroll
  for (int c = 0; c < CHAN; ++c) { float v = hp[c]; ss += v * ap[c]; dd += v * bp[c]; }
  s[i] = ss; d[i] = dd;
}

__device__ __forceinline__ void edge_endpoints(const int* __restrict__ ei, int E, int e,
                                               int& src, int& dst) {
  if (e < E) { src = ei[e]; dst = ei[E + e]; }
  else       { src = dst = e - E; }           // self-loops appended
}

// phase 1: per-dest running max of leaky-relu(score) (stable softmax)
__global__ void k_edge_max(const int* __restrict__ ei, int E, int N,
                           const float* __restrict__ s, const float* __restrict__ d,
                           float* __restrict__ m) {
  int e = blockIdx.x * blockDim.x + threadIdx.x;
  if (e >= E + N) return;
  int src, dst; edge_endpoints(ei, E, e, src, dst);
#pragma unroll
  for (int j = 0; j < NHEAD; ++j) {
    float ev = lrelu(s[src * NHEAD + j] + d[dst * NHEAD + j]);
    atomicMaxFloat(&m[dst * NHEAD + j], ev);
  }
}

// phase 2: per-dest sum of exp(e - max)
__global__ void k_edge_denom(const int* __restrict__ ei, int E, int N,
                             const float* __restrict__ s, const float* __restrict__ d,
                             const float* __restrict__ m, float* __restrict__ den) {
  int e = blockIdx.x * blockDim.x + threadIdx.x;
  if (e >= E + N) return;
  int src, dst; edge_endpoints(ei, E, e, src, dst);
#pragma unroll
  for (int j = 0; j < NHEAD; ++j) {
    float ev = lrelu(s[src * NHEAD + j] + d[dst * NHEAD + j]);
    atomicAdd(&den[dst * NHEAD + j], __expf(ev - m[dst * NHEAD + j]));
  }
}

// phase 3: wave-per-edge weighted scatter-add of h[src] into out[dst]
// lane = channel within head -> fully coalesced 128B loads/atomics per head
__global__ __launch_bounds__(256)
void k_edge_agg(const int* __restrict__ ei, int E, int N,
                const float* __restrict__ s, const float* __restrict__ d,
                const float* __restrict__ m, const float* __restrict__ den,
                const float* __restrict__ h, float* __restrict__ out) {
  int wv   = blockIdx.x * (blockDim.x >> 5) + (threadIdx.x >> 5);
  int lane = threadIdx.x & 31;
  if (wv >= E + N) return;
  int src, dst; edge_endpoints(ei, E, wv, src, dst);
  const float* hs = h + (size_t)src * HCW;
  float* od = out + (size_t)dst * HCW;
#pragma unroll
  for (int j = 0; j < NHEAD; ++j) {
    float ev = lrelu(s[src * NHEAD + j] + d[dst * NHEAD + j]);    // broadcast loads
    float alpha = __expf(ev - m[dst * NHEAD + j]) / (den[dst * NHEAD + j] + 1e-16f);
    atomicAdd(od + j * CHAN + lane, hs[j * CHAN + lane] * alpha);
  }
}

// out = ELU(out + bias), in place
__global__ void k_bias_elu(float* __restrict__ out, const float* __restrict__ b, long total) {
  long i = (long)blockIdx.x * blockDim.x + threadIdx.x;
  if (i >= total) return;
  float v = out[i] + b[i & (HCW - 1)];
  out[i] = v > 0.f ? v : __expf(v) - 1.f;
}

// ---------------- pooling + classifier ----------------
__global__ void k_pool(const float* __restrict__ x, const int* __restrict__ batch,
                       float* __restrict__ pooled, int N) {
  long i = (long)blockIdx.x * blockDim.x + threadIdx.x;
  if (i >= (long)N * HCW) return;
  int n = (int)(i >> 8), c = (int)(i & (HCW - 1));
  atomicAdd(&pooled[(size_t)batch[n] * HCW + c], x[i]);
}

__global__ void k_count(const int* __restrict__ batch, float* __restrict__ cnt, int N) {
  int n = blockIdx.x * blockDim.x + threadIdx.x;
  if (n < N) atomicAdd(&cnt[batch[n]], 1.0f);
}

__global__ void k_final(const float* __restrict__ pooled, const float* __restrict__ cnt,
                        const float* __restrict__ lw, const float* __restrict__ lb,
                        float* __restrict__ y, int G) {
  int i = blockIdx.x * blockDim.x + threadIdx.x;
  if (i >= G * NOUT) return;
  int g = i / NOUT, o = i % NOUT;
  float inv = 1.0f / fmaxf(cnt[g], 1.0f);
  float acc = lb[o];
  for (int c = 0; c < HCW; ++c)
    acc += (pooled[(size_t)g * HCW + c] * inv) * lw[c * NOUT + o];
  y[i] = acc;
}

// ---------------- host orchestration ----------------
extern "C" void kernel_launch(void* const* d_in, const int* in_sizes, int n_in,
                              void* d_out, int out_size, void* d_ws, size_t ws_size,
                              hipStream_t stream) {
  const float* x   = (const float*)d_in[0];
  const int*   ei  = (const int*)d_in[1];
  const int*   bat = (const int*)d_in[2];
  const float* W[3]  = {(const float*)d_in[3],  (const float*)d_in[7],  (const float*)d_in[11]};
  const float* AS[3] = {(const float*)d_in[4],  (const float*)d_in[8],  (const float*)d_in[12]};
  const float* AD[3] = {(const float*)d_in[5],  (const float*)d_in[9],  (const float*)d_in[13]};
  const float* B[3]  = {(const float*)d_in[6],  (const float*)d_in[10], (const float*)d_in[14]};
  const float* lw = (const float*)d_in[15];
  const float* lb = (const float*)d_in[16];
  float* y = (float*)d_out;

  const int N  = in_sizes[2];            // batch vector length == #nodes
  const int E  = in_sizes[1] / 2;        // edge_index is [2,E]
  const int IN = in_sizes[0] / N;        // 128
  const int G  = out_size / NOUT;        // 64
  const int Et = E + N;                  // with self-loops

  // carve workspace
  char* ws = (char*)d_ws;
  size_t off = 0;
  auto carve = [&](size_t bytes) { char* p = ws + off; off += (bytes + 255) & ~(size_t)255; return p; };
  float*    hbuf   = (float*)   carve((size_t)N * HCW * 4);
  float*    buf0   = (float*)   carve((size_t)N * HCW * 4);
  float*    buf1   = (float*)   carve((size_t)N * HCW * 4);
  _Float16* xh     = (_Float16*)carve((size_t)N * HCW * 2);
  _Float16* Wt     = (_Float16*)carve((size_t)HCW * HCW * 2);
  float*    sbuf   = (float*)   carve((size_t)N * NHEAD * 4);
  float*    dbuf   = (float*)   carve((size_t)N * NHEAD * 4);
  float*    mbuf   = (float*)   carve((size_t)N * NHEAD * 4);
  float*    den    = (float*)   carve((size_t)N * NHEAD * 4);
  float*    pooled = (float*)   carve((size_t)G * HCW * 4);
  float*    cnt    = (float*)   carve((size_t)G * 4);
  (void)ws_size; (void)n_in;

  const int T = 256;
  auto blk = [&](long n) { return (unsigned)((n + T - 1) / T); };

  const float* xsrc = x;
  for (int l = 0; l < 3; ++l) {
    const int K = (l == 0) ? IN : HCW;
    float* outb = (l == 1) ? buf1 : buf0;

    // 1) fp16 input + transposed fp16 weights
    k_convert_half<<<blk((long)N * K), T, 0, stream>>>(xsrc, xh, (long)N * K);
    k_transpose_half<<<blk((long)K * HCW), T, 0, stream>>>(W[l], Wt, K);

    // 2) h = x @ W via WMMA
    k_gemm_wmma<<<(N + 15) / 16, 128, 0, stream>>>(xh, Wt, hbuf, N, K);

    // 3) attention scores
    k_scores<<<blk((long)N * NHEAD), T, 0, stream>>>(hbuf, AS[l], AD[l], sbuf, dbuf, N);

    // 4) init segment buffers + output accumulator
    k_fill<<<blk((long)N * NHEAD), T, 0, stream>>>(mbuf, -1e30f, (long)N * NHEAD);
    k_fill<<<blk((long)N * NHEAD), T, 0, stream>>>(den, 0.0f, (long)N * NHEAD);
    k_fill<<<blk((long)N * HCW), T, 0, stream>>>(outb, 0.0f, (long)N * HCW);

    // 5) softmax-weighted aggregation over edges (incl. self-loops)
    k_edge_max  <<<blk(Et), T, 0, stream>>>(ei, E, N, sbuf, dbuf, mbuf);
    k_edge_denom<<<blk(Et), T, 0, stream>>>(ei, E, N, sbuf, dbuf, mbuf, den);
    k_edge_agg  <<<(unsigned)((Et + 7) / 8), 256, 0, stream>>>(ei, E, N, sbuf, dbuf, mbuf, den, hbuf, outb);

    // 6) bias + ELU (in place); output becomes next layer's input
    k_bias_elu<<<blk((long)N * HCW), T, 0, stream>>>(outb, B[l], (long)N * HCW);
    xsrc = outb;
  }

  // mean pool per graph + classifier
  k_fill<<<blk((long)G * HCW), T, 0, stream>>>(pooled, 0.0f, (long)G * HCW);
  k_fill<<<blk(G), T, 0, stream>>>(cnt, 0.0f, G);
  k_pool<<<blk((long)N * HCW), T, 0, stream>>>(xsrc, bat, pooled, N);
  k_count<<<blk(N), T, 0, stream>>>(bat, cnt, N);
  k_final<<<blk((long)G * NOUT), T, 0, stream>>>(pooled, cnt, lw, lb, y, G);
}
